// KPConvLayer_26688926777503
// MI455X (gfx1250) — compile-verified, matrix operand order checked by reading, not run on hardware
//
#include <hip/hip_runtime.h>

// KPConv fused kernel for MI455X (gfx1250, wave32, WMMA bf16 16x16x32).
// N=65536 nodes, K=32 neighbors, Kp=15 kernel points, Cin=64, Cout=128.

typedef __attribute__((ext_vector_type(16))) __bf16 v16bf;
typedef __attribute__((ext_vector_type(8)))  __bf16 v8bf;
typedef __attribute__((ext_vector_type(2)))  __bf16 v2bf;
typedef __attribute__((ext_vector_type(8)))  float  v8f;
typedef __attribute__((ext_vector_type(2)))  float  v2f;

#define SIGMA_INV 10.0f   // 1 / 0.1

// native f32 -> bf16 (lowers to v_cvt_*_bf16_f32 on gfx1250)
__device__ __forceinline__ unsigned short f2bf(float f) {
    return __builtin_bit_cast(unsigned short, (__bf16)f);
}
// packed pair: two f32 -> one dword of 2x bf16 (v_cvt_pk_bf16_f32)
__device__ __forceinline__ unsigned f2bf2(float a, float b) {
    v2f x = {a, b};
    v2bf h = __builtin_convertvector(x, v2bf);
    return __builtin_bit_cast(unsigned, h);
}

__device__ __forceinline__ v16bf cat8(v8bf lo, v8bf hi) {
    return __builtin_shufflevector(lo, hi, 0,1,2,3,4,5,6,7,8,9,10,11,12,13,14,15);
}

// ---------------------------------------------------------------------------
// Prepack weights [Kp*Cin=960, Cout=128] f32 -> bf16 in exact per-lane WMMA
// B-operand layout, so the main kernel loads B as one 32B vector per lane.
// d_ws layout: [t(30)][coutTile(8)][lane(32)][e(16)] bf16  (245760 B)
// element e of lane L in chunk t:  Kchunk = (e<8 ? e : e+8) + 8*(L>=16)
// ---------------------------------------------------------------------------
__global__ void kpconv_prepack(const float* __restrict__ W,
                               unsigned short* __restrict__ ws) {
    int idx = blockIdx.x * 256 + threadIdx.x;
    if (idx >= 30 * 8 * 32 * 16) return;
    int e    = idx & 15;
    int lane = (idx >> 4) & 31;
    int tc   = idx >> 9;          // t*8 + coutTile
    int ct   = tc & 7;
    int t    = tc >> 3;
    int hi   = lane >> 4;
    int kk   = ((e & 8) << 1) + (e & 7) + (hi << 3);  // (e<8?e:e+8)+hi*8
    int K    = t * 32 + kk;                            // 0..959 = kp*64+cin
    int cout = ct * 16 + (lane & 15);
    ws[idx] = f2bf(W[K * 128 + cout]);
}

// ---------------------------------------------------------------------------
// Main fused kernel: 16 nodes per block, 256 threads (8 waves).
//   phase 1: gather neighbor idx/pos, compute infl (bf16, LDS [node][kp][j])
//            gather neighbor feats  (bf16, LDS [node][cin][j], j-pair packed)
//   phase 2: weighted[node][kp][cin] via WMMA  A=infl(16x32) B=feat(32x16)
//   phase 3: out[node][cout] via WMMA over K=960, B prepacked in d_ws
// ---------------------------------------------------------------------------
__global__ __launch_bounds__(256, 2)
void kpconv_main(const float* __restrict__ pos,
                 const float* __restrict__ feats,
                 const float* __restrict__ kpts,
                 const long long* __restrict__ neighbors,
                 const unsigned short* __restrict__ wpack,
                 float* __restrict__ out) {
    __shared__ __align__(16) unsigned short s_infl[16 * 16 * 32];   // [node][kp][j]   16 KB
    __shared__ __align__(16) unsigned short s_feat[16 * 64 * 32];   // [node][cin][j]  64 KB
    __shared__ __align__(16) unsigned short s_wgt [16 * 976];       // [node][K<960]   31 KB
    __shared__ int   s_nbr[512];                                     // [node][j]
    __shared__ float s_kp[48];

    const int tid      = threadIdx.x;
    const int lane     = tid & 31;
    const int wave     = tid >> 5;
    const int laneM    = lane & 15;
    const int hi       = lane >> 4;        // lane group (0: lanes 0-15, 1: 16-31)
    const int nodeBase = blockIdx.x * 16;

    // ---- phase 1a: neighbor indices + kernel points ----
    for (int i = tid; i < 512; i += 256) {
        int node = i >> 5, j = i & 31;
        s_nbr[i] = (int)neighbors[(long long)(nodeBase + node) * 32 + j];
    }
    if (tid < 45) s_kp[tid] = kpts[tid];
    __syncthreads();

    // ---- phase 1b: influence weights ----
    for (int i = tid; i < 512; i += 256) {
        int node = i >> 5, j = i & 31;
        int nb = s_nbr[i];
        float cx = pos[(nodeBase + node) * 3 + 0];
        float cy = pos[(nodeBase + node) * 3 + 1];
        float cz = pos[(nodeBase + node) * 3 + 2];
        float rx = pos[nb * 3 + 0] - cx;
        float ry = pos[nb * 3 + 1] - cy;
        float rz = pos[nb * 3 + 2] - cz;
        #pragma unroll
        for (int kp = 0; kp < 15; ++kp) {
            float dx = rx - s_kp[kp * 3 + 0];
            float dy = ry - s_kp[kp * 3 + 1];
            float dz = rz - s_kp[kp * 3 + 2];
            float dist = sqrtf(dx * dx + dy * dy + dz * dz);
            float infl = fmaxf(0.0f, 1.0f - dist * SIGMA_INV);
            s_infl[node * 512 + kp * 32 + j] = f2bf(infl);
        }
        s_infl[node * 512 + 15 * 32 + j] = 0;   // zero-pad kp row 15
    }

    // ---- phase 1c: gather neighbor feats, j-pair packed bf16 stores ----
    // task = (node, j-pair p, cin-quad q): 16*16*16 = 4096 tasks
    for (int task = tid; task < 4096; task += 256) {
        int q    = task & 15;
        int p    = (task >> 4) & 15;
        int node = task >> 8;
        int r    = node * 32 + p * 2;
        const float4 fa = *(const float4*)(feats + (long long)s_nbr[r]     * 64 + q * 4);
        const float4 fb = *(const float4*)(feats + (long long)s_nbr[r + 1] * 64 + q * 4);
        unsigned* dst = (unsigned*)&s_feat[node * 2048 + (q * 4) * 32 + p * 2];
        dst[ 0] = f2bf2(fa.x, fb.x);    // cin = 4q+0, j = 2p, 2p+1
        dst[16] = f2bf2(fa.y, fb.y);    // cin = 4q+1  (+32 ushorts)
        dst[32] = f2bf2(fa.z, fb.z);    // cin = 4q+2
        dst[48] = f2bf2(fa.w, fb.w);    // cin = 4q+3
    }
    __syncthreads();

    // ---- phase 2: weighted[n,kp,c] = sum_j infl[n,j,kp] * feat[n,j,c] ----
    // One WMMA per (node, cinTile): A = infl (M=kp, K=j=32), B = feat (K=j, N=cin)
    for (int t = wave; t < 64; t += 8) {
        int node = t >> 2;
        int ct   = t & 3;
        int cin  = ct * 16 + laneM;

        v8bf a0 = *(const v8bf*)&s_infl[node * 512 + laneM * 32 + (hi << 3)];
        v8bf a1 = *(const v8bf*)&s_infl[node * 512 + laneM * 32 + 16 + (hi << 3)];
        v16bf a = cat8(a0, a1);

        v8bf b0 = *(const v8bf*)&s_feat[node * 2048 + cin * 32 + (hi << 3)];
        v8bf b1 = *(const v8bf*)&s_feat[node * 2048 + cin * 32 + 16 + (hi << 3)];
        v16bf b = cat8(b0, b1);

        v8f c = {};
        c = __builtin_amdgcn_wmma_f32_16x16x32_bf16(false, a, false, b,
                                                    (short)0, c, false, false);
        #pragma unroll
        for (int v = 0; v < 8; ++v) {
            int kp = v + (hi << 3);             // D: VGPR v -> M = v + 8*hi
            if (kp < 15)
                s_wgt[node * 976 + kp * 64 + cin] = f2bf(c[v]);
        }
    }
    __syncthreads();

    // ---- phase 3: out[n,d] = sum_K weighted[n,K] * W[K,d], K = 960 ----
    // wave -> cout tile (8 tiles of 16); 30 K-chunks of 32
    {
        const int ct = wave;
        v8f acc = {};
        #pragma unroll 2
        for (int t = 0; t < 30; ++t) {
            v8bf a0 = *(const v8bf*)&s_wgt[laneM * 976 + t * 32 + (hi << 3)];
            v8bf a1 = *(const v8bf*)&s_wgt[laneM * 976 + t * 32 + 16 + (hi << 3)];
            v16bf a = cat8(a0, a1);
            // prepacked B: one 32B vector per lane, fully coalesced
            v16bf b = *((const v16bf*)wpack + (t * 8 + ct) * 32 + lane);
            acc = __builtin_amdgcn_wmma_f32_16x16x32_bf16(false, a, false, b,
                                                          (short)0, acc, false, false);
        }
        #pragma unroll
        for (int v = 0; v < 8; ++v) {
            int node = nodeBase + v + (hi << 3);
            out[(long long)node * 128 + ct * 16 + laneM] = acc[v];
        }
    }
}

// ---------------------------------------------------------------------------
extern "C" void kernel_launch(void* const* d_in, const int* in_sizes, int n_in,
                              void* d_out, int out_size, void* d_ws, size_t ws_size,
                              hipStream_t stream) {
    const float*     pos       = (const float*)d_in[0];      // [N,3]
    const float*     feats     = (const float*)d_in[1];      // [N,64]
    const float*     kpts      = (const float*)d_in[2];      // [15,3]
    const float*     weights   = (const float*)d_in[3];      // [15,64,128]
    const long long* neighbors = (const long long*)d_in[4];  // [N,32] int64
    float*           out       = (float*)d_out;              // [N,128]
    unsigned short*  wpack     = (unsigned short*)d_ws;      // 245760 B bf16

    const int N = in_sizes[0] / 3;          // 65536

    kpconv_prepack<<<(30 * 8 * 32 * 16 + 255) / 256, 256, 0, stream>>>(weights, wpack);
    kpconv_main<<<N / 16, 256, 0, stream>>>(pos, feats, kpts, neighbors, wpack, out);
}
